// InfiniAttn_9019431322034
// MI455X (gfx1250) — matrix operand order repaired and checked by reading, not compiled, not run on hardware
//
#include <hip/hip_runtime.h>
#include <hip/hip_bf16.h>

// ---------------------------------------------------------------------------
// InfiniAttn forward for MI455X (gfx1250), bf16 WMMA everywhere.
//   S=2048 B=2 D=1024 N=16 DH=64
// Requires ws_size >= 64 MB.
// Flash-attention stages K/V tiles into LDS with GLOBAL_LOAD_ASYNC_TO_LDS_B128
// double-buffered on ASYNCcnt.
// ---------------------------------------------------------------------------

typedef unsigned short u16;
typedef unsigned int   u32;
typedef __bf16 bf16x16 __attribute__((ext_vector_type(16)));
typedef float  v8f     __attribute__((ext_vector_type(8)));

#define S_   2048
#define B_   2
#define D_   1024
#define NH_  16
#define DH_  64
#define M_   (S_ * B_)   // 4096 rows of "token" matrices

// ----------------------------- helpers -------------------------------------

__device__ inline u16 f2bf(float x) {            // round-to-nearest-even
    u32 u = __float_as_uint(x);
    u32 r = u + 0x7fffu + ((u >> 16) & 1u);
    return (u16)(r >> 16);
}
__device__ inline float bf2f(u16 b) { return __uint_as_float(((u32)b) << 16); }

// A-matrix fragment (16x32 bf16), ISA 7.12.2 layout:
// lane = {g = lane>>4, r = lane&15}; row = r;
// VGPR t holds K pair at k = (t<4 ? 2t : 16+2(t-4)) + 8g
__device__ inline bf16x16 load_a_frag(const u16* rowptr, int g) {
    union { bf16x16 v; u32 u[8]; } f;
#pragma unroll
    for (int t = 0; t < 8; ++t) {
        int k = ((t < 4) ? (2 * t) : (16 + 2 * (t - 4))) + 8 * g;
        f.u[t] = *(const u32*)(rowptr + k);
    }
    return f.v;
}

// B-matrix fragment (32x16 bf16): lane holds column n = lane&15; element s at
// k = (s<8 ? s : 16+(s-8)) + 8*(lane>>4), gathered with row stride ld.
__device__ inline bf16x16 load_b_frag(const u16* colptr, int ld, int g) {
    union { bf16x16 v; u16 h[16]; } f;
#pragma unroll
    for (int s = 0; s < 16; ++s) {
        int k = ((s < 8) ? s : (16 + (s - 8))) + 8 * g;
        f.h[s] = colptr[(size_t)k * ld];
    }
    return f.v;
}

__device__ inline v8f wmma_bf16(bf16x16 a, bf16x16 b, v8f c) {
    return __builtin_amdgcn_wmma_f32_16x16x32_bf16(false, a, false, b,
                                                   (short)0, c, false, false);
}

// ---- gfx1250 async Global->LDS DMA (ASYNCcnt-tracked) ----------------------
// Generic LDS pointers carry the workgroup-relative LDS offset in their low
// 32 bits (flat aperture rule), which is exactly what VDST of
// GLOBAL_LOAD_ASYNC_TO_LDS wants.
__device__ inline u32 lds_off(const void* p) { return (u32)(size_t)p; }

__device__ inline void async_ld_b128(u32 lds_byte_off, const void* gaddr) {
    asm volatile("global_load_async_to_lds_b128 %0, %1, off"
                 :: "v"(lds_byte_off), "v"(gaddr)
                 : "memory");
}
__device__ inline void wait_async_0() {
#if __has_builtin(__builtin_amdgcn_s_wait_asynccnt)
    __builtin_amdgcn_s_wait_asynccnt(0);
#else
    asm volatile("s_wait_asynccnt 0x0" ::: "memory");
#endif
}
__device__ inline void wait_async_4() {
#if __has_builtin(__builtin_amdgcn_s_wait_asynccnt)
    __builtin_amdgcn_s_wait_asynccnt(4);
#else
    asm volatile("s_wait_asynccnt 0x4" ::: "memory");
#endif
}

// ----------------------------- kernels -------------------------------------

__global__ void k_f32_to_bf16(const float* __restrict__ in,
                              u16* __restrict__ out, int n) {
    int i = blockIdx.x * blockDim.x + threadIdx.x;
    if (i < n) out[i] = f2bf(in[i]);
}

// Kt[b][n][d][s] = K[s,b,n,d]  (K = first half of KV columns)
__global__ void k_transpose_k(const u16* __restrict__ kv, u16* __restrict__ kt) {
    int idx = blockIdx.x * blockDim.x + threadIdx.x; // over B*NH*DH*S
    int s = idx & (S_ - 1);
    int d = (idx >> 11) & 63;
    int n = (idx >> 17) & 15;
    int b = idx >> 21;
    kt[idx] = kv[((size_t)s * B_ + b) * (2 * D_) + n * DH_ + d];
}

// C(bf16)[M x N] = A(bf16)[M x K] @ B(bf16)[K x N]; grid = (N/128, M/64), 256 thr
__global__ __launch_bounds__(256)
void k_gemm_bf16(const u16* __restrict__ A, const u16* __restrict__ Bm,
                 u16* __restrict__ C, int K, int N) {
    const int lane = threadIdx.x & 31, wv = threadIdx.x >> 5;
    const int r = lane & 15, g = lane >> 4;
    const int row0 = blockIdx.y * 64 + (wv & 1) * 32;
    const int col0 = blockIdx.x * 128 + (wv >> 1) * 32;
    const v8f z = {0.f,0.f,0.f,0.f,0.f,0.f,0.f,0.f};
    v8f acc[2][2] = {{z, z}, {z, z}};
    const u16* arow = A + (size_t)(row0 + r) * K;
    for (int kk = 0; kk < K; kk += 32) {
        if (kk + 32 < K) {
            __builtin_prefetch(arow + kk + 32, 0, 1);             // global_prefetch_b8
        }
        bf16x16 a0 = load_a_frag(arow + kk, g);
        bf16x16 a1 = load_a_frag(arow + (size_t)16 * K + kk, g);
        const u16* brow = Bm + (size_t)kk * N + col0 + r;
        bf16x16 b0 = load_b_frag(brow, N, g);
        bf16x16 b1 = load_b_frag(brow + 16, N, g);
        acc[0][0] = wmma_bf16(a0, b0, acc[0][0]);
        acc[0][1] = wmma_bf16(a0, b1, acc[0][1]);
        acc[1][0] = wmma_bf16(a1, b0, acc[1][0]);
        acc[1][1] = wmma_bf16(a1, b1, acc[1][1]);
    }
#pragma unroll
    for (int tm = 0; tm < 2; ++tm)
#pragma unroll
        for (int tn = 0; tn < 2; ++tn)
#pragma unroll
            for (int e = 0; e < 8; ++e) {
                int row = row0 + tm * 16 + e + 8 * g;
                int col = col0 + tn * 16 + r;
                C[(size_t)row * N + col] = f2bf(acc[tm][tn][e]);
            }
}

// out(f32) = A(bf16) @ B(bf16) + resid(f32); same tiling as k_gemm_bf16
__global__ __launch_bounds__(256)
void k_gemm_out(const u16* __restrict__ A, const u16* __restrict__ Bm,
                const float* __restrict__ resid, float* __restrict__ out,
                int K, int N) {
    const int lane = threadIdx.x & 31, wv = threadIdx.x >> 5;
    const int r = lane & 15, g = lane >> 4;
    const int row0 = blockIdx.y * 64 + (wv & 1) * 32;
    const int col0 = blockIdx.x * 128 + (wv >> 1) * 32;
    const v8f z = {0.f,0.f,0.f,0.f,0.f,0.f,0.f,0.f};
    v8f acc[2][2] = {{z, z}, {z, z}};
    const u16* arow = A + (size_t)(row0 + r) * K;
    for (int kk = 0; kk < K; kk += 32) {
        bf16x16 a0 = load_a_frag(arow + kk, g);
        bf16x16 a1 = load_a_frag(arow + (size_t)16 * K + kk, g);
        const u16* brow = Bm + (size_t)kk * N + col0 + r;
        bf16x16 b0 = load_b_frag(brow, N, g);
        bf16x16 b1 = load_b_frag(brow + 16, N, g);
        acc[0][0] = wmma_bf16(a0, b0, acc[0][0]);
        acc[0][1] = wmma_bf16(a0, b1, acc[0][1]);
        acc[1][0] = wmma_bf16(a1, b0, acc[1][0]);
        acc[1][1] = wmma_bf16(a1, b1, acc[1][1]);
    }
#pragma unroll
    for (int tm = 0; tm < 2; ++tm)
#pragma unroll
        for (int tn = 0; tn < 2; ++tn)
#pragma unroll
            for (int e = 0; e < 8; ++e) {
                int row = row0 + tm * 16 + e + 8 * g;
                int col = col0 + tn * 16 + r;
                size_t o = (size_t)row * N + col;
                out[o] = acc[tm][tn][e] + resid[o];
            }
}

// Flash attention: 4 waves per WG, each wave owns a 16-row query tile.
// K/V j-tiles staged in LDS via async DMA, double-buffered on ASYNCcnt.
// grid = (S/64, B*NH), 128 threads.
__global__ __launch_bounds__(128)
void k_flash_attn(const u16* __restrict__ Qb, const u16* __restrict__ KVb,
                  const u16* __restrict__ Kt, u16* __restrict__ attnb) {
    __shared__ u16 KTs[2][64 * 32];   // [d][j] tile, 4 KB each
    __shared__ u16 VTs[2][32 * 64];   // [j][d] tile, 4 KB each
    __shared__ u16 PTs[4][16 * 32];   // per-wave P tile

    const int tid  = threadIdx.x;
    const int lane = tid & 31, wv = tid >> 5;
    const int r = lane & 15, g = lane >> 4;
    const int head = blockIdx.y;          // head = b*NH + n
    const int b = head >> 4;
    const int n = head & 15;
    const int i0 = blockIdx.x * 64 + wv * 16;
    const float scale = 0.125f;           // 1/sqrt(DH)

    const u16* ktbase = Kt + (size_t)head * DH_ * S_;
    const u16* vbase  = KVb + (size_t)b * (2 * D_) + D_ + n * DH_; // + j*B_*2D_

    // Q fragments (16 rows x 64 cols), per-wave
    const u16* qrow = Qb + ((size_t)(i0 + r) * B_ + b) * D_ + n * DH_;
    bf16x16 qa0 = load_a_frag(qrow, g);       // d = 0..31
    bf16x16 qa1 = load_a_frag(qrow + 32, g);  // d = 32..63

    float m[8], l[8], corr[8];
    const v8f z = {0.f,0.f,0.f,0.f,0.f,0.f,0.f,0.f};
    v8f O[4] = {z, z, z, z};
#pragma unroll
    for (int e = 0; e < 8; ++e) { m[e] = -1e30f; l[e] = 0.f; }

    // issue one (K,V) j-tile: 4 async b128 per thread = 8 KB across 128 thr
    auto issue = [&](int buf, int j0) {
#pragma unroll
        for (int q = 0; q < 2; ++q) {     // K tile: [64 x 32] elems
            int idx = (q * 128 + tid) * 8;
            int d = idx >> 5, js = idx & 31;
            async_ld_b128(lds_off(&KTs[buf][idx]),
                          ktbase + (size_t)d * S_ + j0 + js);
        }
#pragma unroll
        for (int q = 0; q < 2; ++q) {     // V tile: [32 x 64] elems
            int idx = (q * 128 + tid) * 8;
            int j = idx >> 6, ds = idx & 63;
            async_ld_b128(lds_off(&VTs[buf][idx]),
                          vbase + (size_t)(j0 + j) * (B_ * 2 * D_) + ds);
        }
    };

    int buf = 0;
    issue(0, 0);
    for (int j0 = 0; j0 < S_; j0 += 32) {
        if (j0 + 32 < S_) {               // prefetch next tile, wait current
            issue(buf ^ 1, j0 + 32);
            wait_async_4();
        } else {
            wait_async_0();
        }
        __syncthreads();                  // whole tile visible to all waves

        // ---- scores: two 16x16 tiles over 32 key columns (K from LDS) ----
        const u16* kr0 = &KTs[buf][r];                        // d 0..31
        const u16* kr1 = &KTs[buf][32 * 32 + r];              // d 32..63
        v8f s0 = z, s1 = z;
        s0 = wmma_bf16(qa0, load_b_frag(kr0, 32, g),      s0);
        s0 = wmma_bf16(qa1, load_b_frag(kr1, 32, g),      s0);
        s1 = wmma_bf16(qa0, load_b_frag(kr0 + 16, 32, g), s1);
        s1 = wmma_bf16(qa1, load_b_frag(kr1 + 16, 32, g), s1);

        // ---- online softmax (row stats land on the right lanes) ----
        u16* Pt = PTs[wv];
#pragma unroll
        for (int e = 0; e < 8; ++e) {
            float a = s0[e] * scale, c = s1[e] * scale;
            float t = fmaxf(a, c);
#pragma unroll
            for (int mk = 1; mk < 16; mk <<= 1)
                t = fmaxf(t, __shfl_xor(t, mk, 32));
            float mn = fmaxf(m[e], t);
            float cr = __expf(m[e] - mn);
            m[e] = mn;
            corr[e] = cr;
            float p0 = __expf(a - mn);
            float p1 = __expf(c - mn);
            float rs = p0 + p1;
#pragma unroll
            for (int mk = 1; mk < 16; mk <<= 1)
                rs += __shfl_xor(rs, mk, 32);
            l[e] = l[e] * cr + rs;
            // P tile to per-wave LDS (C layout): row = e + 8g, cols r / r+16
            Pt[(e + 8 * g) * 32 + r]      = f2bf(p0);
            Pt[(e + 8 * g) * 32 + r + 16] = f2bf(p1);
        }
#pragma unroll
        for (int t4 = 0; t4 < 4; ++t4)
#pragma unroll
            for (int e = 0; e < 8; ++e) O[t4][e] *= corr[e];

        // reload P as a 16x32 A fragment (same-wave LDS ops are in-order)
        union { bf16x16 v; u32 u[8]; } pf;
#pragma unroll
        for (int t = 0; t < 8; ++t) {
            int k = ((t < 4) ? (2 * t) : (16 + 2 * (t - 4))) + 8 * g;
            pf.u[t] = *(const u32*)&Pt[r * 32 + k];
        }

        // ---- O += P @ V  (V tile from LDS) ----
        const u16* vrow = &VTs[buf][r];
#pragma unroll
        for (int t4 = 0; t4 < 4; ++t4)
            O[t4] = wmma_bf16(pf.v, load_b_frag(vrow + t4 * 16, 64, g), O[t4]);

        __syncthreads();                  // everyone done with buf before reuse
        buf ^= 1;
    }

    // normalize + store attn_vec (bf16) at [s,b,n,d]
#pragma unroll
    for (int t4 = 0; t4 < 4; ++t4)
#pragma unroll
        for (int e = 0; e < 8; ++e) {
            int row = i0 + e + 8 * g;
            int col = t4 * 16 + r;
            attnb[((size_t)row * B_ + b) * D_ + n * DH_ + col] =
                f2bf(O[t4][e] / l[e]);
        }
}

// compressive-memory readout + gated injection; one block per (s,b,n), 64 thr
__global__ __launch_bounds__(64)
void k_content_gate(const u16* __restrict__ Qb, const u16* __restrict__ attnb,
                    const float* __restrict__ memory,
                    const float* __restrict__ mem_norm,
                    const float* __restrict__ gate_beta,
                    u16* __restrict__ injb) {
    int idx = blockIdx.x;
    int n = idx & 15, b = (idx >> 4) & 1, s = idx >> 5;
    int t = threadIdx.x;
    size_t off = ((size_t)s * B_ + b) * D_ + n * DH_;

    float qv = bf2f(Qb[off + t]);
    float cq = qv > 0.f ? qv + 1.f : __expf(qv);   // elu(q)+1

    __shared__ float cqs[64];
    __shared__ float red[64];
    cqs[t] = cq;
    red[t] = cq * mem_norm[((size_t)b * NH_ + n) * DH_ + t];
    __syncthreads();
    for (int o = 32; o > 0; o >>= 1) {
        if (t < o) red[t] += red[t + o];
        __syncthreads();
    }
    float den = red[0];

    const float* Mh = memory + ((size_t)b * NH_ + n) * DH_ * DH_;
    float num = 0.f;
#pragma unroll 8
    for (int k = 0; k < DH_; ++k) num = fmaf(cqs[k], Mh[k * DH_ + t], num);

    float gb = gate_beta[0];
    float gg = 1.f / (1.f + __expf(-gb));
    float av = bf2f(attnb[off + t]);
    injb[off + t] = f2bf(gg * (num / den) + (1.f - gg) * av);
}

// in-place LayerNorm over last dim (1024); one block per row
__global__ __launch_bounds__(256)
void k_layernorm(float* __restrict__ y, const float* __restrict__ gamma,
                 const float* __restrict__ beta) {
    size_t base = (size_t)blockIdx.x * D_;
    int t = threadIdx.x;
    float v[4], s = 0.f, s2 = 0.f;
#pragma unroll
    for (int i = 0; i < 4; ++i) {
        v[i] = y[base + t + i * 256];
        s += v[i];
        s2 += v[i] * v[i];
    }
    __shared__ float r1[256], r2[256];
    r1[t] = s; r2[t] = s2;
    __syncthreads();
    for (int o = 128; o > 0; o >>= 1) {
        if (t < o) { r1[t] += r1[t + o]; r2[t] += r2[t + o]; }
        __syncthreads();
    }
    float mu = r1[0] * (1.f / D_);
    float var = r2[0] * (1.f / D_) - mu * mu;
    float rstd = rsqrtf(var + 1e-5f);
#pragma unroll
    for (int i = 0; i < 4; ++i) {
        int c = t + i * 256;
        y[base + c] = (v[i] - mu) * rstd * gamma[c] + beta[c];
    }
}

// ----------------------------- launch --------------------------------------

extern "C" void kernel_launch(void* const* d_in, const int* in_sizes, int n_in,
                              void* d_out, int out_size, void* d_ws, size_t ws_size,
                              hipStream_t stream) {
    const float* h        = (const float*)d_in[0];
    const float* Wq       = (const float*)d_in[1];
    const float* Wkv      = (const float*)d_in[2];
    const float* Wo       = (const float*)d_in[3];
    const float* gamma    = (const float*)d_in[4];
    const float* beta     = (const float*)d_in[5];
    const float* memory   = (const float*)d_in[6];
    const float* mem_norm = (const float*)d_in[7];
    const float* gate_b   = (const float*)d_in[8];

    char* ws = (char*)d_ws;                         // needs >= 64 MB
    const size_t SZ_HB  = (size_t)M_ * D_ * 2;      // 8 MB  bf16 h
    const size_t SZ_WQ  = (size_t)D_ * D_ * 2;      // 2 MB
    const size_t SZ_WKV = (size_t)D_ * 2 * D_ * 2;  // 4 MB
    const size_t SZ_WO  = (size_t)D_ * D_ * 2;      // 2 MB
    const size_t SZ_Q   = (size_t)M_ * D_ * 2;      // 8 MB
    const size_t SZ_KV  = (size_t)M_ * 2 * D_ * 2;  // 16 MB
    const size_t SZ_KT  = (size_t)B_ * NH_ * DH_ * S_ * 2; // 8 MB
    const size_t SZ_AT  = (size_t)M_ * D_ * 2;      // 8 MB
    u16* hb   = (u16*)(ws);
    u16* wqb  = (u16*)(ws + SZ_HB);
    u16* wkvb = (u16*)(ws + SZ_HB + SZ_WQ);
    u16* wob  = (u16*)(ws + SZ_HB + SZ_WQ + SZ_WKV);
    u16* Qb   = (u16*)(ws + SZ_HB + SZ_WQ + SZ_WKV + SZ_WO);
    u16* KVb  = (u16*)(ws + SZ_HB + SZ_WQ + SZ_WKV + SZ_WO + SZ_Q);
    u16* Kt   = (u16*)(ws + SZ_HB + SZ_WQ + SZ_WKV + SZ_WO + SZ_Q + SZ_KV);
    u16* attnb= (u16*)(ws + SZ_HB + SZ_WQ + SZ_WKV + SZ_WO + SZ_Q + SZ_KV + SZ_KT);
    u16* injb = (u16*)(ws + SZ_HB + SZ_WQ + SZ_WKV + SZ_WO + SZ_Q + SZ_KV + SZ_KT + SZ_AT);

    // 1) fp32 -> bf16 conversions
    k_f32_to_bf16<<<(M_ * D_) / 256, 256, 0, stream>>>(h, hb, M_ * D_);
    k_f32_to_bf16<<<(D_ * D_) / 256, 256, 0, stream>>>(Wq, wqb, D_ * D_);
    k_f32_to_bf16<<<(D_ * 2 * D_) / 256, 256, 0, stream>>>(Wkv, wkvb, D_ * 2 * D_);
    k_f32_to_bf16<<<(D_ * D_) / 256, 256, 0, stream>>>(Wo, wob, D_ * D_);

    // 2) projections (WMMA GEMMs)
    k_gemm_bf16<<<dim3(D_ / 128, M_ / 64), 256, 0, stream>>>(hb, wqb, Qb, D_, D_);
    k_gemm_bf16<<<dim3((2 * D_) / 128, M_ / 64), 256, 0, stream>>>(hb, wkvb, KVb, D_, 2 * D_);

    // 3) per-head K transpose -> [B,NH,DH,S]
    k_transpose_k<<<(B_ * NH_ * DH_ * S_) / 256, 256, 0, stream>>>(KVb, Kt);

    // 4) flash attention (WMMA QK^T and PV; async K/V tiles into LDS)
    k_flash_attn<<<dim3(S_ / 64, B_ * NH_), 128, 0, stream>>>(Qb, KVb, Kt, attnb);

    // 5) compressive-memory readout + gate -> inj (bf16)
    k_content_gate<<<S_ * B_ * NH_, 64, 0, stream>>>(Qb, attnb, memory, mem_norm,
                                                     gate_b, injb);

    // 6) output projection + residual (WMMA GEMM, fp32 out)
    k_gemm_out<<<dim3(D_ / 128, M_ / 64), 256, 0, stream>>>(injb, wob, h,
                                                            (float*)d_out, D_, D_);

    // 7) in-place LayerNorm on d_out
    k_layernorm<<<M_, 256, 0, stream>>>((float*)d_out, gamma, beta);
}
